// FraudGAT_10952166605400
// MI455X (gfx1250) — matrix operand order repaired and checked by reading, not compiled, not run on hardware
//
#include <hip/hip_runtime.h>
#include <hip/hip_bf16.h>

// ---------------------------------------------------------------------------
// FraudGAT forward for gfx1250 (MI455X). WMMA bf16 path for all GEMMs.
// Weights pre-packed per-launch into bf16 in the WMMA B-lane layout.
// ---------------------------------------------------------------------------

#define N_NODES 50000
#define N_EDGES 800000
#define NODE_IN 128
#define EDGE_IN 16
#define HID     64
#define HEADS   2
#define NEG_SLOPE 0.2f

typedef __attribute__((ext_vector_type(16))) __bf16 v16bf;
typedef __attribute__((ext_vector_type(8)))  float  v8f;

__device__ __forceinline__ v8f wmma_bf16(v16bf a, v16bf b, v8f c) {
  // (neg_a, A, neg_b, B, c_mod, C, reuse_a, reuse_b)
  return __builtin_amdgcn_wmma_f32_16x16x32_bf16(false, a, false, b,
                                                 (short)0, c, false, false);
}

// monotonic float -> uint key for atomicMax-based segment max
__device__ __forceinline__ unsigned fkey(float f) {
  unsigned u = __float_as_uint(f);
  return (u & 0x80000000u) ? ~u : (u | 0x80000000u);
}
__device__ __forceinline__ float funkey(unsigned k) {
  return __uint_as_float((k & 0x80000000u) ? (k & 0x7FFFFFFFu) : ~k);
}

__device__ __forceinline__ float elu1(float x) {
  return x > 0.0f ? x : (__expf(x) - 1.0f);
}

// ---------------------------------------------------------------------------
// zero init (accumulators, denominators, max-keys)
// ---------------------------------------------------------------------------
__global__ void k_zero32(unsigned* __restrict__ p, size_t n) {
  size_t i = blockIdx.x * (size_t)blockDim.x + threadIdx.x;
  size_t stride = (size_t)gridDim.x * blockDim.x;
  for (; i < n; i += stride) p[i] = 0u;
}

// ---------------------------------------------------------------------------
// Pack B[K,N] (f32, row-major) -> bf16 in per-lane WMMA B layout.
// Vector slot i = (kt*ntiles + nt)*32 + lane holds that lane's 16 bf16 K-pair
// values for K-step kt, N-tile nt.  K is the padded K (mult of 32); rows
// >= Kreal are zero-filled (classifier K=144 -> pad 160).
// ---------------------------------------------------------------------------
__global__ void k_packB_bf16(const float* __restrict__ B, __bf16* __restrict__ Bp,
                             int Kpad, int Kreal, int N) {
  const int ntiles = N >> 4;
  const int total  = (Kpad >> 5) * ntiles * 32;
  int i = blockIdx.x * blockDim.x + threadIdx.x;
  if (i >= total) return;
  const int lane = i & 31;
  const int tile = i >> 5;
  const int nt = tile % ntiles, kt = tile / ntiles;
  const int half = lane >> 4, l16 = lane & 15;
  const int ncol = nt * 16 + l16;
  const int kb   = kt * 32 + half * 8;    // lanes0-15: K 0-7,16-23; lanes16-31: K 8-15,24-31
  v16bf v;
#pragma unroll
  for (int r = 0; r < 8; ++r) {
    const int k0 = kb + r, k1 = kb + 16 + r;
    v[r]     = (__bf16)((k0 < Kreal) ? B[(size_t)k0 * N + ncol] : 0.0f);
    v[8 + r] = (__bf16)((k1 < Kreal) ? B[(size_t)k1 * N + ncol] : 0.0f);
  }
  ((v16bf*)Bp)[i] = v;
}

// ---------------------------------------------------------------------------
// weff[k][h] = sum_c We[k][h*C+c] * att_e[h][c]   (fold edge GEMM -> 16-dot)
// ---------------------------------------------------------------------------
__global__ void k_weff(const float* __restrict__ We, const float* __restrict__ att_e,
                       float* __restrict__ weff, int Kin, int heads, int C) {
  int i = blockIdx.x * blockDim.x + threadIdx.x;
  if (i >= Kin * heads) return;
  int k = i / heads, h = i % heads;
  float s = 0.0f;
  for (int c = 0; c < C; ++c)
    s += We[(size_t)k * heads * C + h * C + c] * att_e[h * C + c];
  weff[k * heads + h] = s;
}

// ---------------------------------------------------------------------------
// bf16 WMMA GEMM with pre-packed B: C[M,N] = A[M,K] @ B[K,N]
// Requires: M%16==0, K%32==0, N%16==0. One wave per 16x16 C tile.
// B operand: one contiguous 32B load per lane per K-step.
// ---------------------------------------------------------------------------
__global__ __launch_bounds__(256)
void k_gemm_bf16(const float* __restrict__ A, const __bf16* __restrict__ Bp,
                 float* __restrict__ C, int M, int K, int N) {
  const int lane = threadIdx.x & 31;
  const int wave = threadIdx.x >> 5;
  const int ntiles = N >> 4;
  const int total = (M >> 4) * ntiles;
  int t = blockIdx.x * (blockDim.x >> 5) + wave;
  if (t >= total) return;
  const int mt = t / ntiles, nt = t % ntiles;
  const int half = lane >> 4;          // K sub-block selector
  const int l16  = lane & 15;
  const int mrow = mt * 16 + l16;      // A row this lane feeds
  const v16bf* Bv = (const v16bf*)Bp;
  v8f acc = {};
  for (int kt = 0; kt < (K >> 5); ++kt) {
    const int k0 = kt << 5;
    // A 16x32 bf16 layout: lanes0-15 K=[k0..k0+7],[k0+16..k0+23]; lanes16-31 +8
    const float* Ap = A + (size_t)mrow * K + k0 + half * 8;
    v16bf a;
#pragma unroll
    for (int j = 0; j < 8; ++j) {
      a[j]     = (__bf16)Ap[j];
      a[8 + j] = (__bf16)Ap[16 + j];
    }
    const v16bf b = Bv[(size_t)(kt * ntiles + nt) * 32 + lane];
    acc = wmma_bf16(a, b, acc);
  }
  // C layout: VGPR r -> row mt*16 + half*8 + r, col = nt*16 + l16
  float* Cp = C + (size_t)(mt * 16 + half * 8) * N + nt * 16 + l16;
#pragma unroll
  for (int r = 0; r < 8; ++r) Cp[(size_t)r * N] = acc[r];
}

// ---------------------------------------------------------------------------
// per-(node,head) attention scalars: a_s = h·att_s, a_d = h·att_d
// ---------------------------------------------------------------------------
__global__ void k_node_att(const float* __restrict__ H, const float* __restrict__ att_s,
                           const float* __restrict__ att_d, float* __restrict__ as,
                           float* __restrict__ ad, int N, int heads, int C) {
  int i = blockIdx.x * blockDim.x + threadIdx.x;
  if (i >= N * heads) return;
  int n = i / heads, h = i % heads;
  const float* hp = H + (size_t)n * heads * C + h * C;
  float s = 0.0f, d = 0.0f;
  for (int c = 0; c < C; ++c) {
    float v = hp[c];
    s += v * att_s[h * C + c];
    d += v * att_d[h * C + c];
  }
  as[i] = s;
  ad[i] = d;
}

// ---------------------------------------------------------------------------
// per-(edge,head): logits = leaky_relu(a_s[src]+a_d[dst]+ea·weff); seg-max(dst)
// ---------------------------------------------------------------------------
__global__ void k_logits_max(const int* __restrict__ src, const int* __restrict__ dst,
                             const float* __restrict__ ea, const float* __restrict__ weff,
                             const float* __restrict__ as, const float* __restrict__ ad,
                             float* __restrict__ logits, unsigned* __restrict__ mkey,
                             int E, int heads) {
  int i = blockIdx.x * blockDim.x + threadIdx.x;
  if (i >= E * heads) return;
  int e = i / heads, h = i % heads;
  const float4* ep = (const float4*)(ea + (size_t)e * EDGE_IN);
  float s = 0.0f;
#pragma unroll
  for (int q = 0; q < EDGE_IN / 4; ++q) {
    float4 v = ep[q];
    s += v.x * weff[(4 * q + 0) * heads + h];
    s += v.y * weff[(4 * q + 1) * heads + h];
    s += v.z * weff[(4 * q + 2) * heads + h];
    s += v.w * weff[(4 * q + 3) * heads + h];
  }
  int S = src[e], D = dst[e];
  float l = as[S * heads + h] + ad[D * heads + h] + s;
  l = l > 0.0f ? l : NEG_SLOPE * l;
  logits[i] = l;
  atomicMax(&mkey[D * heads + h], fkey(l));
}

// ---------------------------------------------------------------------------
// ex = exp(logit - max[dst]); overwrite logits with ex; seg-sum denom(dst)
// ---------------------------------------------------------------------------
__global__ void k_exp_sum(const int* __restrict__ dst, float* __restrict__ logits,
                          const unsigned* __restrict__ mkey, float* __restrict__ denom,
                          int E, int heads) {
  int i = blockIdx.x * blockDim.x + threadIdx.x;
  if (i >= E * heads) return;
  int e = i / heads, h = i % heads;
  int D = dst[e];
  float m  = funkey(mkey[D * heads + h]);
  float ex = __expf(logits[i] - m);
  logits[i] = ex;
  atomicAdd(&denom[D * heads + h], ex);
}

// ---------------------------------------------------------------------------
// out[dst, f..f+3] += H[src, f..f+3] * ex/(denom[dst]+eps)   (float4 gather)
// ---------------------------------------------------------------------------
__global__ void k_scatter4(const int* __restrict__ src, const int* __restrict__ dst,
                           const float* __restrict__ H, const float* __restrict__ ex,
                           const float* __restrict__ denom, float* __restrict__ out,
                           int E, int heads, int C) {
  const int F  = heads * C;
  const int F4 = F >> 2;
  size_t i = blockIdx.x * (size_t)blockDim.x + threadIdx.x;
  if (i >= (size_t)E * F4) return;
  int e = (int)(i / F4), q = (int)(i % F4);
  int f = q << 2, h = f / C;                      // C%4==0 -> same head for all 4
  int S = src[e], D = dst[e];
  float alpha = ex[(size_t)e * heads + h] / (denom[D * heads + h] + 1e-16f);
  const float4 hv = *(const float4*)(H + (size_t)S * F + f);
  float* op = out + (size_t)D * F + f;
  atomicAdd(op + 0, hv.x * alpha);
  atomicAdd(op + 1, hv.y * alpha);
  atomicAdd(op + 2, hv.z * alpha);
  atomicAdd(op + 3, hv.w * alpha);
}

// ---------------------------------------------------------------------------
// out = elu(out + bias[f])  in place
// ---------------------------------------------------------------------------
__global__ void k_bias_elu(float* __restrict__ out, const float* __restrict__ bias,
                           int N, int F) {
  size_t i = blockIdx.x * (size_t)blockDim.x + threadIdx.x;
  if (i >= (size_t)N * F) return;
  out[i] = elu1(out[i] + bias[i % F]);
}

// ---------------------------------------------------------------------------
// Fused edge classifier: out[e] = relu([h2[src]|h2[dst]|ea] @ Wc1 + bc1) @ Wc2 + bc2
// Wave computes 16 edges; K=144 padded to 160; 4 N-tiles of bf16 WMMA against
// pre-packed Wc1; hid never materialized. Shuffle-reduce over the N lane group.
// ---------------------------------------------------------------------------
__device__ __forceinline__ float edge_repr_elem(const float* __restrict__ h2,
                                                const float* __restrict__ ea,
                                                int srow, int drow, int e, int c) {
  if (c < 64)  return h2[(size_t)srow * HID + c];
  if (c < 128) return h2[(size_t)drow * HID + (c - 64)];
  if (c < 144) return ea[(size_t)e * EDGE_IN + (c - 128)];
  return 0.0f;
}

__global__ __launch_bounds__(256)
void k_edge_mlp(const float* __restrict__ h2, const int* __restrict__ src,
                const int* __restrict__ dst, const float* __restrict__ ea,
                const __bf16* __restrict__ Wc1p, const float* __restrict__ bc1,
                const float* __restrict__ Wc2, const float* __restrict__ bc2,
                float* __restrict__ out, int E) {
  const int lane = threadIdx.x & 31;
  const int wave = threadIdx.x >> 5;
  int t = blockIdx.x * (blockDim.x >> 5) + wave;
  if (t >= (E >> 4)) return;
  const int eb   = t << 4;            // 16 edges per wave
  const int half = lane >> 4;
  const int l16  = lane & 15;
  const int eL   = eb + l16;          // this lane's A row (edge)
  const int srow = src[eL], drow = dst[eL];
  const v16bf* Bv = (const v16bf*)Wc1p;   // packed [kt][nt][lane], ntiles=4

  float partial[8];
#pragma unroll
  for (int r = 0; r < 8; ++r) partial[r] = 0.0f;

  for (int nt = 0; nt < 4; ++nt) {        // N = 64 -> 4 tiles of 16
    const int ncol = nt * 16 + l16;
    v8f acc = {};
#pragma unroll
    for (int kt = 0; kt < 5; ++kt) {      // K = 144 padded to 160
      const int ka = kt * 32 + half * 8;
      v16bf a;
#pragma unroll
      for (int j = 0; j < 8; ++j) {
        a[j]     = (__bf16)edge_repr_elem(h2, ea, srow, drow, eL, ka + j);
        a[8 + j] = (__bf16)edge_repr_elem(h2, ea, srow, drow, eL, ka + 16 + j);
      }
      const v16bf b = Bv[(size_t)(kt * 4 + nt) * 32 + lane];
      acc = wmma_bf16(a, b, acc);
    }
    // hid = relu(acc + bc1[ncol]); contract with Wc2[ncol]
    const float w = Wc2[ncol], bb = bc1[ncol];
#pragma unroll
    for (int r = 0; r < 8; ++r) partial[r] += fmaxf(acc[r] + bb, 0.0f) * w;
  }
  // reduce over the 16-lane N group (xor of lane bits 0..3 stays in-half)
#pragma unroll
  for (int r = 0; r < 8; ++r) {
#pragma unroll
    for (int m = 1; m < 16; m <<= 1) partial[r] += __shfl_xor(partial[r], m, 32);
  }
  if (l16 == 0) {
    const float b2 = bc2[0];
#pragma unroll
    for (int r = 0; r < 8; ++r) out[eb + half * 8 + r] = partial[r] + b2;
  }
}

// ---------------------------------------------------------------------------
// host-side orchestration
// ---------------------------------------------------------------------------
static inline unsigned cdiv_u(size_t a, unsigned b) { return (unsigned)((a + b - 1) / b); }

extern "C" void kernel_launch(void* const* d_in, const int* in_sizes, int n_in,
                              void* d_out, int out_size, void* d_ws, size_t ws_size,
                              hipStream_t stream) {
  (void)in_sizes; (void)n_in; (void)out_size; (void)ws_size;
  const float* x       = (const float*)d_in[0];
  const int*   eidx    = (const int*)  d_in[1];
  const float* ea      = (const float*)d_in[2];
  const float* W1      = (const float*)d_in[3];
  const float* att_s1  = (const float*)d_in[4];
  const float* att_d1  = (const float*)d_in[5];
  const float* We1     = (const float*)d_in[6];
  const float* att_e1  = (const float*)d_in[7];
  const float* b1      = (const float*)d_in[8];
  const float* W2      = (const float*)d_in[9];
  const float* att_s2  = (const float*)d_in[10];
  const float* att_d2  = (const float*)d_in[11];
  const float* We2     = (const float*)d_in[12];
  const float* att_e2  = (const float*)d_in[13];
  const float* b2      = (const float*)d_in[14];
  const float* Wc1     = (const float*)d_in[15];
  const float* bc1     = (const float*)d_in[16];
  const float* Wc2     = (const float*)d_in[17];
  const float* bc2     = (const float*)d_in[18];
  float* out = (float*)d_out;

  const int* src = eidx;
  const int* dst = eidx + N_EDGES;

  // workspace carving
  size_t off = 0;
  auto carve = [&](size_t bytes) -> char* {
    size_t o = (off + 255) & ~(size_t)255;
    off = o + bytes;
    return (char*)d_ws + o;
  };
  float*    H1    = (float*)   carve((size_t)N_NODES * HEADS * HID * 4);
  float*    O1    = (float*)   carve((size_t)N_NODES * HEADS * HID * 4);
  float*    AS1   = (float*)   carve((size_t)N_NODES * HEADS * 4);
  float*    AD1   = (float*)   carve((size_t)N_NODES * HEADS * 4);
  float*    WEFF1 = (float*)   carve((size_t)EDGE_IN * HEADS * 4);
  float*    LOG1  = (float*)   carve((size_t)N_EDGES * HEADS * 4);
  unsigned* MK1   = (unsigned*)carve((size_t)N_NODES * HEADS * 4);
  float*    DEN1  = (float*)   carve((size_t)N_NODES * HEADS * 4);
  float*    H2    = (float*)   carve((size_t)N_NODES * HID * 4);
  float*    O2    = (float*)   carve((size_t)N_NODES * HID * 4);
  float*    AS2   = (float*)   carve((size_t)N_NODES * 4);
  float*    AD2   = (float*)   carve((size_t)N_NODES * 4);
  float*    WEFF2 = (float*)   carve((size_t)EDGE_IN * 4);
  float*    LOG2  = (float*)   carve((size_t)N_EDGES * 4);
  unsigned* MK2   = (unsigned*)carve((size_t)N_NODES * 4);
  float*    DEN2  = (float*)   carve((size_t)N_NODES * 4);
  // packed bf16 weights (B-lane layout)
  __bf16*   W1P   = (__bf16*)  carve((size_t)(NODE_IN / 32) * (HEADS * HID / 16) * 32 * 16 * 2);
  __bf16*   W2P   = (__bf16*)  carve((size_t)(HEADS * HID / 32) * (HID / 16) * 32 * 16 * 2);
  __bf16*   WC1P  = (__bf16*)  carve((size_t)(160 / 32) * (HID / 16) * 32 * 16 * 2);

  const unsigned B = 256;

  // ---- init accumulators (per-call for graph-replay determinism) ----
  k_zero32<<<2048, B, 0, stream>>>((unsigned*)O1,  (size_t)N_NODES * HEADS * HID);
  k_zero32<<<256,  B, 0, stream>>>(MK1,            (size_t)N_NODES * HEADS);
  k_zero32<<<256,  B, 0, stream>>>((unsigned*)DEN1,(size_t)N_NODES * HEADS);
  k_zero32<<<2048, B, 0, stream>>>((unsigned*)O2,  (size_t)N_NODES * HID);
  k_zero32<<<256,  B, 0, stream>>>(MK2,            (size_t)N_NODES);
  k_zero32<<<256,  B, 0, stream>>>((unsigned*)DEN2,(size_t)N_NODES);

  // ---- pack weights to bf16 B-lane layout ----
  k_packB_bf16<<<cdiv_u((NODE_IN / 32) * (HEADS * HID / 16) * 32, B), B, 0, stream>>>(
      W1, W1P, NODE_IN, NODE_IN, HEADS * HID);
  k_packB_bf16<<<cdiv_u((HEADS * HID / 32) * (HID / 16) * 32, B), B, 0, stream>>>(
      W2, W2P, HEADS * HID, HEADS * HID, HID);
  k_packB_bf16<<<cdiv_u((160 / 32) * (HID / 16) * 32, B), B, 0, stream>>>(
      Wc1, WC1P, 160, 144, HID);

  // ---- conv1 ----
  {
    const int M = N_NODES, K = NODE_IN, N = HEADS * HID;    // 50000 x 128 x 128
    unsigned tiles = (unsigned)((M / 16) * (N / 16));
    k_gemm_bf16<<<cdiv_u(tiles, 8), B, 0, stream>>>(x, W1P, H1, M, K, N);
  }
  k_weff<<<1, 64, 0, stream>>>(We1, att_e1, WEFF1, EDGE_IN, HEADS, HID);
  k_node_att<<<cdiv_u((size_t)N_NODES * HEADS, B), B, 0, stream>>>(
      H1, att_s1, att_d1, AS1, AD1, N_NODES, HEADS, HID);
  k_logits_max<<<cdiv_u((size_t)N_EDGES * HEADS, B), B, 0, stream>>>(
      src, dst, ea, WEFF1, AS1, AD1, LOG1, MK1, N_EDGES, HEADS);
  k_exp_sum<<<cdiv_u((size_t)N_EDGES * HEADS, B), B, 0, stream>>>(
      dst, LOG1, MK1, DEN1, N_EDGES, HEADS);
  k_scatter4<<<cdiv_u((size_t)N_EDGES * HEADS * HID / 4, B), B, 0, stream>>>(
      src, dst, H1, LOG1, DEN1, O1, N_EDGES, HEADS, HID);
  k_bias_elu<<<cdiv_u((size_t)N_NODES * HEADS * HID, B), B, 0, stream>>>(
      O1, b1, N_NODES, HEADS * HID);

  // ---- conv2 (heads = 1) ----
  {
    const int M = N_NODES, K = HEADS * HID, N = HID;        // 50000 x 128 x 64
    unsigned tiles = (unsigned)((M / 16) * (N / 16));
    k_gemm_bf16<<<cdiv_u(tiles, 8), B, 0, stream>>>(O1, W2P, H2, M, K, N);
  }
  k_weff<<<1, 64, 0, stream>>>(We2, att_e2, WEFF2, EDGE_IN, 1, HID);
  k_node_att<<<cdiv_u((size_t)N_NODES, B), B, 0, stream>>>(
      H2, att_s2, att_d2, AS2, AD2, N_NODES, 1, HID);
  k_logits_max<<<cdiv_u((size_t)N_EDGES, B), B, 0, stream>>>(
      src, dst, ea, WEFF2, AS2, AD2, LOG2, MK2, N_EDGES, 1);
  k_exp_sum<<<cdiv_u((size_t)N_EDGES, B), B, 0, stream>>>(
      dst, LOG2, MK2, DEN2, N_EDGES, 1);
  k_scatter4<<<cdiv_u((size_t)N_EDGES * HID / 4, B), B, 0, stream>>>(
      src, dst, H2, LOG2, DEN2, O2, N_EDGES, 1, HID);
  k_bias_elu<<<cdiv_u((size_t)N_NODES * HID, B), B, 0, stream>>>(
      O2, b2, N_NODES, HID);

  // ---- fused edge classifier (WMMA) ----
  k_edge_mlp<<<cdiv_u(N_EDGES / 16, 8), B, 0, stream>>>(
      O2, src, dst, ea, WC1P, bc1, Wc2, bc2, out, N_EDGES);
}
